// MyMamba_31739808317885
// MI455X (gfx1250) — matrix operand order repaired and checked by reading, not compiled, not run on hardware
//
#include <hip/hip_runtime.h>
#include <hip/hip_bf16.h>
#include <math.h>

// ---------------------------------------------------------------------------
// Types for CDNA5 WMMA
// ---------------------------------------------------------------------------
typedef __attribute__((ext_vector_type(16))) __bf16 v16bf;
typedef __attribute__((ext_vector_type(8)))  float  v8f;

#define NSEQ   1024         // BATCH * N_CH
#define LPAT   64           // PATCH_NUM
#define DM     128          // D_MODEL
#define NL     (NSEQ*LPAT)  // 65536 rows
#define NLD    ((size_t)NL*DM)

__device__ __forceinline__ float gelu_exact(float x) {
    return 0.5f * x * (1.0f + erff(x * 0.70710678118654752f));
}
__device__ __forceinline__ float silu(float x) {
    return x / (1.0f + __expf(-x));
}
__device__ __forceinline__ float softplus_f(float x) {
    return (x > 20.0f) ? x : log1pf(__expf(x));
}

__device__ __forceinline__ void cvt8(v16bf& dst, int base, float4 x, float4 y,
                                     float s) {
    dst[base + 0] = (__bf16)(x.x * s); dst[base + 1] = (__bf16)(x.y * s);
    dst[base + 2] = (__bf16)(x.z * s); dst[base + 3] = (__bf16)(x.w * s);
    dst[base + 4] = (__bf16)(y.x * s); dst[base + 5] = (__bf16)(y.y * s);
    dst[base + 6] = (__bf16)(y.z * s); dst[base + 7] = (__bf16)(y.w * s);
}

// ---------------------------------------------------------------------------
// Generic bf16-WMMA GEMM:  C[M,N] = act( A[M,K] * W[N,K]^T [+ bias] [+ add] )
// Compile-time HAS_BIAS / HAS_ADD / ACT keep each epilogue straight-line.
// One wave computes one 16x16 tile. REQUIRES: M % 16 == 0, K % 32 == 0,
// lda/ldw % 4 == 0 (true for all call sites). All fragment loads are
// unconditional b128; invalid columns (N edge) zeroed by a 0/1 scale.
// ACT: 0=none, 1=gelu
// ---------------------------------------------------------------------------
template <bool HAS_BIAS, bool HAS_ADD, int ACT>
__global__ void gemm_bf16_wmma(const float* __restrict__ A, int lda,
                               const float* __restrict__ W, int ldw,
                               const float* __restrict__ bias,
                               const float* __restrict__ add, int ldadd,
                               float* __restrict__ C, int ldc,
                               int N, int K) {
    const int lane = threadIdx.x & 31;
    const int n0 = blockIdx.x * 16;
    const int m0 = blockIdx.y * 16;

    const int arow = m0 + (lane & 15);
    const int kba  = (lane & 16) ? 16 : 0;   // A K-base per half-wave (contiguous)
    const int col  = n0 + (lane & 15);
    const int kbb  = (lane & 16) ? 16 : 0;   // B K-base per half-wave
    const bool colv = col < N;
    const float bscale = colv ? 1.0f : 0.0f; // branch-free invalid-column zeroing

    const float* ap = A + (size_t)arow * lda;
    const float* wp = W + (size_t)(colv ? col : 0) * ldw;

    v8f acc = {};
    for (int k0 = 0; k0 < K; k0 += 32) {
        const float4* a_lo = (const float4*)(ap + k0 + kba);        // K kb..kb+7
        const float4* a_hi = (const float4*)(ap + k0 + kba + 8);    // K kb+8..kb+15
        const float4* bp   = (const float4*)(wp + k0 + kbb);        // 16 contiguous
        float4 al0 = a_lo[0], al1 = a_lo[1];
        float4 ah0 = a_hi[0], ah1 = a_hi[1];
        float4 b0 = bp[0], b1 = bp[1], b2 = bp[2], b3 = bp[3];
        v16bf a, b;
        cvt8(a, 0, al0, al1, 1.0f);
        cvt8(a, 8, ah0, ah1, 1.0f);
        cvt8(b, 0, b0, b1, bscale);
        cvt8(b, 8, b2, b3, bscale);
        acc = __builtin_amdgcn_wmma_f32_16x16x32_bf16(false, a, false, b,
                                                      (short)0, acc, false, false);
    }

    if (!colv) return;                       // divergence only after last WMMA
    float bv = 0.0f;
    if (HAS_BIAS) bv = bias[col];
    const int rbase = m0 + ((lane & 16) ? 8 : 0);
#pragma unroll
    for (int v = 0; v < 8; ++v) {
        int row = rbase + v;
        float val = acc[v] + bv;
        if (HAS_ADD) val += add[(size_t)row * ldadd + col];
        if (ACT == 1) val = gelu_exact(val);
        C[(size_t)row * ldc + col] = val;
    }
}

// ---------------------------------------------------------------------------
// fc1 with fused patch extraction:
//   p[n,l,k] = x[b, min(l*8+k,511), c]   (n = b*32+c)
//   h = gelu(p @ fc1_w^T + fc1_b)        K=16 padded into one 16x16x32 WMMA
// ---------------------------------------------------------------------------
__global__ void fc1_patch_kernel(const float* __restrict__ x,
                                 const float* __restrict__ w,   // (128,16)
                                 const float* __restrict__ bias,
                                 float* __restrict__ out) {      // (NL,128)
    const int lane = threadIdx.x & 31;
    const int n0 = blockIdx.x * 16;
    const int m0 = blockIdx.y * 16;

    const int row  = m0 + (lane & 15);
    const int nseq = row >> 6;               // sequence 0..1023
    const int l    = row & 63;               // patch index
    const int bi   = nseq >> 5;              // batch
    const int ci   = nseq & 31;              // channel
    const int kba  = (lane & 16) ? 8 : 0;

    v16bf a, b;
#pragma unroll
    for (int i = 0; i < 8; ++i) {
        int k = kba + i;                     // 0..15, always valid (K=16)
        int s = l * 8 + k; if (s > 511) s = 511;   // right-pad with last sample
        a[i]     = (__bf16)x[((size_t)bi * 512 + s) * 32 + ci];
        a[8 + i] = (__bf16)0.0f;             // K 16..31 = zero pad
    }
    const int col = n0 + (lane & 15);
    {   // lanes 0-15 carry K 0..7, lanes 16-31 carry K 8..15 in slots 0..7; slots 8..15 (K 16..31) = 0
        const float4* bp = (const float4*)(w + col * 16 + ((lane & 16) ? 8 : 0));
        float4 b0 = bp[0], b1 = bp[1];
        cvt8(b, 0, b0, b1, 1.0f);
        cvt8(b, 8, b0, b1, 0.0f);
    }

    v8f acc = {};
    acc = __builtin_amdgcn_wmma_f32_16x16x32_bf16(false, a, false, b,
                                                  (short)0, acc, false, false);
    const float bv = bias[col];
    const int rbase = m0 + ((lane & 16) ? 8 : 0);
#pragma unroll
    for (int v = 0; v < 8; ++v) {
        float val = gelu_exact(acc[v] + bv);
        out[(size_t)(rbase + v) * DM + col] = val;
    }
}

// ---------------------------------------------------------------------------
// Delta projection (K=8, one WMMA):
//   delta = softplus(dbl[:, :8] @ dtw^T + dtb)
// Fragment sub-blocks carrying K>=8 are zeroed with a 0/1 half-wave scale;
// the 8 valid floats come from two unconditional float4 loads.
// ---------------------------------------------------------------------------
__global__ void dt_gemm_kernel(const float* __restrict__ dbl,   // (NL,40)
                               const float* __restrict__ dtw,   // (128,8)
                               const float* __restrict__ dtb,   // (128)
                               float* __restrict__ delta) {     // (NL,128)
    const int lane = threadIdx.x & 31;
    const int n0 = blockIdx.x * 16;
    const int m0 = blockIdx.y * 16;

    const int row = m0 + (lane & 15);
    const int col = n0 + (lane & 15);
    const float slo = (lane & 16) ? 0.0f : 1.0f;  // upper half-wave K-slots -> 0

    v16bf a, b;
    {   // A: lanes 0-15 carry K 0..7 (valid), lanes 16-31 carry K 8..15 (zero)
        const float4* ap = (const float4*)(dbl + (size_t)row * 40);  // 160B-aligned
        float4 a0 = ap[0], a1 = ap[1];
        cvt8(a, 0, a0, a1, slo);
#pragma unroll
        for (int i = 8; i < 16; ++i) a[i] = (__bf16)0.0f;  // K 16..31 = 0
    }
    {   // B: lanes 0-15 carry K 0..15 (K 0..7 valid), lanes 16-31 carry K 16..31
        const float4* bp = (const float4*)(dtw + col * 8);           // 32B-aligned
        float4 b0 = bp[0], b1 = bp[1];
        cvt8(b, 0, b0, b1, slo);
#pragma unroll
        for (int j = 8; j < 16; ++j) b[j] = (__bf16)0.0f;  // K 8..15 / 24..31 = 0
    }

    v8f acc = {};
    acc = __builtin_amdgcn_wmma_f32_16x16x32_bf16(false, a, false, b,
                                                  (short)0, acc, false, false);
    const float bv = dtb[col];
    const int rbase = m0 + ((lane & 16) ? 8 : 0);
#pragma unroll
    for (int v = 0; v < 8; ++v) {
        delta[(size_t)(rbase + v) * DM + col] = softplus_f(acc[v] + bv);
    }
}

// ---------------------------------------------------------------------------
// BatchNorm over axes (N, D) per patch index l : mean/rstd
// ---------------------------------------------------------------------------
__global__ void bn_stats_kernel(const float* __restrict__ h,
                                float* __restrict__ stats) {     // (64,2)
    const int l = blockIdx.x;
    __shared__ float s_sum[256], s_sq[256];
    float sum = 0.0f, sq = 0.0f;
    for (int i = threadIdx.x; i < NSEQ * DM; i += 256) {
        int n = i >> 7, d = i & 127;
        float v = h[((size_t)n * LPAT + l) * DM + d];
        sum += v; sq += v * v;
    }
    s_sum[threadIdx.x] = sum; s_sq[threadIdx.x] = sq;
    __syncthreads();
    for (int off = 128; off > 0; off >>= 1) {
        if (threadIdx.x < off) {
            s_sum[threadIdx.x] += s_sum[threadIdx.x + off];
            s_sq[threadIdx.x]  += s_sq[threadIdx.x + off];
        }
        __syncthreads();
    }
    if (threadIdx.x == 0) {
        const float inv = 1.0f / (float)(NSEQ * DM);
        float mean = s_sum[0] * inv;
        float var  = s_sq[0] * inv - mean * mean;
        stats[l * 2]     = mean;
        stats[l * 2 + 1] = rsqrtf(var + 1e-5f);
    }
}

__global__ void bn_apply_kernel(float* __restrict__ h,
                                const float* __restrict__ stats,
                                const float* __restrict__ g,
                                const float* __restrict__ b) {
    size_t idx = (size_t)blockIdx.x * blockDim.x + threadIdx.x;
    if (idx >= NLD) return;
    int l = (int)((idx >> 7) & 63);
    h[idx] = (h[idx] - stats[l * 2]) * stats[l * 2 + 1] * g[l] + b[l];
}

// ---------------------------------------------------------------------------
// Depthwise causal conv (kernel 2) + SiLU:  u = silu(u_{l-1}*w0 + u_l*w1 + b)
// reads channel d from xz[..., 0:128]
// ---------------------------------------------------------------------------
__global__ void conv_silu_kernel(const float* __restrict__ xz,   // (NL,256)
                                 const float* __restrict__ cw,   // (128,2)
                                 const float* __restrict__ cb,
                                 float* __restrict__ u) {        // (NL,128)
    size_t idx = (size_t)blockIdx.x * blockDim.x + threadIdx.x;
    if (idx >= NLD) return;
    int d = (int)(idx & 127);
    size_t row = idx >> 7;                   // n*64 + l
    int l = (int)(row & 63);
    float cur  = xz[row * 256 + d];
    float prev = l ? xz[(row - 1) * 256 + d] : 0.0f;
    float v = prev * cw[d * 2] + cur * cw[d * 2 + 1] + cb[d];
    u[idx] = silu(v);
}

// ---------------------------------------------------------------------------
// Selective scan: one block per sequence, one thread per channel d.
// 16-entry state in registers; B/C (32 floats) staged in LDS per timestep.
// ---------------------------------------------------------------------------
__global__ void scan_kernel(const float* __restrict__ u,       // (NL,128)
                            const float* __restrict__ delta,   // (NL,128)
                            const float* __restrict__ dbl,     // (NL,40): [dt|B|C]
                            const float* __restrict__ Alog,    // (128,16)
                            const float* __restrict__ Dp,      // (128)
                            float* __restrict__ y) {           // (NL,128)
    const int n = blockIdx.x;
    const int d = threadIdx.x;
    __shared__ float Bs[16], Cs[16];

    float A[16];
#pragma unroll
    for (int s = 0; s < 16; ++s) A[s] = -__expf(Alog[d * 16 + s]);
    const float Dd = Dp[d];
    float hst[16];
#pragma unroll
    for (int s = 0; s < 16; ++s) hst[s] = 0.0f;

    for (int l = 0; l < LPAT; ++l) {
        size_t row = (size_t)n * LPAT + l;
        if (threadIdx.x < 32) {
            const float* bc = dbl + row * 40 + 8;   // B then C
            if (threadIdx.x < 16) Bs[threadIdx.x] = bc[threadIdx.x];
            else                  Cs[threadIdx.x - 16] = bc[threadIdx.x];
        }
        __syncthreads();
        float uv = u[row * DM + d];
        float dt = delta[row * DM + d];
        float du = dt * uv;
        float acc = 0.0f;
#pragma unroll
        for (int s = 0; s < 16; ++s) {
            hst[s] = __expf(dt * A[s]) * hst[s] + du * Bs[s];
            acc += hst[s] * Cs[s];
        }
        y[row * DM + d] = acc + uv * Dd;
        __syncthreads();
    }
}

// y *= silu(z),  z = xz[..., 128:256]
__global__ void gate_kernel(float* __restrict__ y, const float* __restrict__ xz) {
    size_t idx = (size_t)blockIdx.x * blockDim.x + threadIdx.x;
    if (idx >= NLD) return;
    int d = (int)(idx & 127);
    size_t row = idx >> 7;
    float z = xz[row * 256 + 128 + d];
    y[idx] *= silu(z);
}

// (1024,96) -> (32,96,32):  out[b,t,c] = in[b*32+c, t]
__global__ void out_transpose_kernel(const float* __restrict__ in,
                                     float* __restrict__ out) {
    int idx = blockIdx.x * blockDim.x + threadIdx.x;
    if (idx >= 32 * 96 * 32) return;
    int c = idx & 31;
    int t = (idx >> 5) % 96;
    int b = idx / (96 * 32);
    out[idx] = in[(size_t)(b * 32 + c) * 96 + t];
}

// ---------------------------------------------------------------------------
// Host side
// ---------------------------------------------------------------------------
static void run_mamba(hipStream_t stream, float* h, const float* res_add,
                      float* xz, float* u, float* dbl, float* delta, float* y,
                      const float* inproj, const float* convw, const float* convb,
                      const float* xproj, const float* dtw, const float* dtb,
                      const float* Alog, const float* Dv, const float* outproj) {
    dim3 w(32);
    // xz = h @ inproj^T                       (65536,128)x(256,128)^T
    gemm_bf16_wmma<false, false, 0><<<dim3(16, NL / 16), w, 0, stream>>>(
        h, DM, inproj, DM, nullptr, nullptr, 0, xz, 256, 256, DM);
    // depthwise conv + silu
    conv_silu_kernel<<<(NLD + 255) / 256, 256, 0, stream>>>(xz, convw, convb, u);
    // dbl = u @ xproj^T                       (65536,128)x(40,128)^T
    gemm_bf16_wmma<false, false, 0><<<dim3(3, NL / 16), w, 0, stream>>>(
        u, DM, xproj, DM, nullptr, nullptr, 0, dbl, 40, 40, DM);
    // delta = softplus(dbl[:, :8] @ dtw^T + dtb)   (specialized K=8 kernel)
    dt_gemm_kernel<<<dim3(8, NL / 16), w, 0, stream>>>(dbl, dtw, dtb, delta);
    // selective scan
    scan_kernel<<<NSEQ, DM, 0, stream>>>(u, delta, dbl, Alog, Dv, y);
    // y *= silu(z)
    gate_kernel<<<(NLD + 255) / 256, 256, 0, stream>>>(y, xz);
    // h = y @ outproj^T (+ residual)
    if (res_add) {
        gemm_bf16_wmma<false, true, 0><<<dim3(8, NL / 16), w, 0, stream>>>(
            y, DM, outproj, DM, nullptr, res_add, DM, h, DM, DM, DM);
    } else {
        gemm_bf16_wmma<false, false, 0><<<dim3(8, NL / 16), w, 0, stream>>>(
            y, DM, outproj, DM, nullptr, nullptr, 0, h, DM, DM, DM);
    }
}

extern "C" void kernel_launch(void* const* d_in, const int* in_sizes, int n_in,
                              void* d_out, int out_size, void* d_ws, size_t ws_size,
                              hipStream_t stream) {
    (void)in_sizes; (void)n_in; (void)out_size; (void)ws_size;
    const float* x      = (const float*)d_in[0];
    const float* fc1_w  = (const float*)d_in[1];
    const float* fc1_b  = (const float*)d_in[2];
    const float* bn_g   = (const float*)d_in[3];
    const float* bn_b   = (const float*)d_in[4];
    const float* fc2_w  = (const float*)d_in[5];
    const float* fc2_b  = (const float*)d_in[6];
    const float* m_p[2][9];
    for (int m = 0; m < 2; ++m)
        for (int j = 0; j < 9; ++j)
            m_p[m][j] = (const float*)d_in[7 + m * 9 + j];
    const float* fc3_w  = (const float*)d_in[25];
    const float* fc3_b  = (const float*)d_in[26];
    const float* fc4_w  = (const float*)d_in[27];
    const float* fc4_b  = (const float*)d_in[28];

    char* ws = (char*)d_ws;
    const size_t HB = NLD * sizeof(float);        // 32 MB
    float* h     = (float*)(ws);
    float* res   = (float*)(ws + HB);
    float* xz    = (float*)(ws + 2 * HB);          // 2*HB wide
    float* u     = (float*)(ws + 4 * HB);
    float* dbl   = (float*)(ws + 5 * HB);          // NL*40 floats (10.5 MB)
    float* delta = (float*)(ws + 5 * HB + 16777216);
    float* y     = (float*)(ws + 6 * HB + 16777216);
    float* stats = (float*)(ws + 7 * HB + 16777216);
    float* fc3o  = (float*)(ws + 7 * HB + 16777216 + 4096);
    float* fc4o  = (float*)(ws + 7 * HB + 16777216 + 4096 + 1048576);

    dim3 w(32);
    // 1) patches -> fc1 -> gelu
    fc1_patch_kernel<<<dim3(8, NL / 16), w, 0, stream>>>(x, fc1_w, fc1_b, h);
    // 2) BatchNorm over (N, D) per patch
    bn_stats_kernel<<<LPAT, 256, 0, stream>>>(h, stats);
    bn_apply_kernel<<<(NLD + 255) / 256, 256, 0, stream>>>(h, stats, bn_g, bn_b);
    // 3) res = h @ fc2^T + fc2_b
    gemm_bf16_wmma<true, false, 0><<<dim3(8, NL / 16), w, 0, stream>>>(
        h, DM, fc2_w, DM, fc2_b, nullptr, 0, res, DM, DM, DM);
    // 4) h = mamba1(h) + res ;  5) h = mamba2(h)
    run_mamba(stream, h, res,     xz, u, dbl, delta, y,
              m_p[0][0], m_p[0][1], m_p[0][2], m_p[0][3], m_p[0][4],
              m_p[0][5], m_p[0][6], m_p[0][7], m_p[0][8]);
    run_mamba(stream, h, nullptr, xz, u, dbl, delta, y,
              m_p[1][0], m_p[1][1], m_p[1][2], m_p[1][3], m_p[1][4],
              m_p[1][5], m_p[1][6], m_p[1][7], m_p[1][8]);
    // 6) fc3: (1024, 8192) @ (192, 8192)^T -> gelu
    gemm_bf16_wmma<true, false, 1><<<dim3(12, 64), w, 0, stream>>>(
        h, LPAT * DM, fc3_w, LPAT * DM, fc3_b, nullptr, 0, fc3o, 192,
        192, LPAT * DM);
    // 7) fc4: (1024, 192) @ (96, 192)^T
    gemm_bf16_wmma<true, false, 0><<<dim3(6, 64), w, 0, stream>>>(
        fc3o, 192, fc4_w, 192, fc4_b, nullptr, 0, fc4o, 96,
        96, 192);
    // 8) (B*C, 96) -> (B, 96, C)
    out_transpose_kernel<<<(32 * 96 * 32 + 255) / 256, 256, 0, stream>>>(
        fc4o, (float*)d_out);
}
